// VectorQuantizerG_46901042873039
// MI455X (gfx1250) — matrix operand order repaired and checked by reading, not compiled, not run on hardware
//
#include <hip/hip_runtime.h>

#define N_ 262144
#define D_ 256
#define K_ 1024
#define G_ 4
#define C_ 64

typedef _Float16 half8  __attribute__((ext_vector_type(8)));
typedef _Float16 half16 __attribute__((ext_vector_type(16)));
typedef float    v8f    __attribute__((ext_vector_type(8)));

static constexpr int TILES = 2;                    // 32-row tiles per wave per iter
static constexpr int ROWS_PER_WG = 8 * 32 * TILES; // 8 waves * 32 rows * 2 = 512
static constexpr int CB_STRIDE = 144;              // bytes per code row in LDS (128 + 16 pad)
// + 4KB slack so the software-pipeline prefetch of (unused) tile 64 stays in-bounds
static constexpr size_t SMEM_BYTES = (size_t)K_ * CB_STRIDE + (size_t)K_ * sizeof(float) + 4096;

__device__ __forceinline__ half16 cat8(half8 a, half8 b) {
    return __builtin_shufflevector(a, b, 0,1,2,3,4,5,6,7,8,9,10,11,12,13,14,15);
}

// ---------------------------------------------------------------------------
// Prep: f32 codebooks -> f16 copy in workspace, e_sq per code, zero loss slot.
// ---------------------------------------------------------------------------
__global__ __launch_bounds__(256) void vq_prep(const float* __restrict__ cb,
                                               _Float16* __restrict__ cbws,
                                               float* __restrict__ esqws,
                                               float* __restrict__ loss) {
    int t = blockIdx.x * 256 + threadIdx.x;   // 0..4095 == g*K + k
    const float* src = cb + (size_t)t * C_;
    _Float16* dst = cbws + (size_t)t * C_;
    float s = 0.f;
#pragma unroll 8
    for (int c = 0; c < C_; ++c) {
        float v = src[c];
        s += v * v;
        dst[c] = (_Float16)v;
    }
    esqws[t] = s;
    if (t == 0) *loss = 0.f;
}

// ---------------------------------------------------------------------------
// Main kernel
// ---------------------------------------------------------------------------
__global__ __launch_bounds__(256) void vq_main(const float* __restrict__ z,
                                               const float* __restrict__ cb,
                                               const _Float16* __restrict__ cbws,
                                               const float* __restrict__ esqws,
                                               float* __restrict__ out,
                                               float* __restrict__ loss) {
    extern __shared__ char smem[];
    const int g   = blockIdx.y;
    const int tid = threadIdx.x;

    // ---- stage group codebook (f16, padded rows) and e_sq into LDS ----
    {
        const uint4* srcb = (const uint4*)(cbws + (size_t)g * K_ * C_);
        for (int n = tid; n < K_; n += 256) {
            uint4* dst = (uint4*)(smem + n * CB_STRIDE);
            const uint4* s8 = srcb + n * 8;
#pragma unroll
            for (int j = 0; j < 8; ++j) dst[j] = s8[j];
        }
        float* sesq = (float*)(smem + (size_t)K_ * CB_STRIDE);
        const float* se = esqws + g * K_;
        for (int i = tid; i < K_; i += 256) sesq[i] = se[i];
    }
    __syncthreads();

    const int lane = tid & 31;
    const int wave = tid >> 5;
    const int lhi  = lane >> 4;   // 0 or 1
    const int llo  = lane & 15;
    const float* sesq = (const float*)(smem + (size_t)K_ * CB_STRIDE);
    const char* pbase = smem + lhi * 32 + (size_t)llo * CB_STRIDE;

    for (int it = 0; it < TILES; ++it) {
        const int row_base = blockIdx.x * ROWS_PER_WG + wave * (32 * TILES) + it * 32;

        // ---- load two A tiles (rows row_base..+15 and +16..+31) ----
        half16 a0[2], a1[2];
        float zsq[2];
#pragma unroll
        for (int s = 0; s < 2; ++s) {
            const float* zrow = z + (size_t)(row_base + s * 16 + llo) * D_ + g * C_ + lhi * 16;
            float zq = 0.f;
#pragma unroll
            for (int run = 0; run < 4; ++run) {   // c-offsets (run>>1)*32 + (run&1)*8; runs 0,1 -> a0; 2,3 -> a1
                const float4* p = (const float4*)(zrow + (run >> 1) * 32 + (run & 1) * 8);
                float4 f0 = p[0], f1 = p[1];
                float v[8] = {f0.x, f0.y, f0.z, f0.w, f1.x, f1.y, f1.z, f1.w};
#pragma unroll
                for (int e = 0; e < 8; ++e) {
                    zq = fmaf(v[e], v[e], zq);
                    _Float16 h = (_Float16)v[e];
                    int j = (run & 1) * 8 + e;
                    if (run < 2) a0[s][j] = h; else a1[s][j] = h;
                }
            }
            zq += __shfl_xor(zq, 16, 32);
            zsq[s] = zq;
        }

        float best[2][8];
        int   bidx[2][8];
#pragma unroll
        for (int s = 0; s < 2; ++s)
#pragma unroll
            for (int i = 0; i < 8; ++i) { best[s][i] = 3.4e38f; bidx[s][i] = 0; }

        // ---- software-pipelined sweep over K=1024 codes (64 tiles of 16) ----
        half16 b0 = cat8(*(const half8*)(pbase),      *(const half8*)(pbase + 16));
        half16 b1 = cat8(*(const half8*)(pbase + 64), *(const half8*)(pbase + 80));
        float  ecur = sesq[llo];

        for (int t = 0; t < 64; ++t) {
            // prefetch tile t+1 (t=63 prefetches in-bounds garbage, never used)
            const char* q = pbase + (size_t)(t + 1) * (16 * CB_STRIDE);
            half16 nb0 = cat8(*(const half8*)(q),      *(const half8*)(q + 16));
            half16 nb1 = cat8(*(const half8*)(q + 64), *(const half8*)(q + 80));
            float  enxt = sesq[(t + 1) * 16 + llo];

            v8f accA = {};
            accA = __builtin_amdgcn_wmma_f32_16x16x32_f16(false, a0[0], false, b0,
                                                          (short)0, accA, false, false);
            v8f accB = {};
            accB = __builtin_amdgcn_wmma_f32_16x16x32_f16(false, a0[1], false, b0,
                                                          (short)0, accB, false, false);
            accA = __builtin_amdgcn_wmma_f32_16x16x32_f16(false, a1[0], false, b1,
                                                          (short)0, accA, false, false);
            accB = __builtin_amdgcn_wmma_f32_16x16x32_f16(false, a1[1], false, b1,
                                                          (short)0, accB, false, false);

            const int cand = t * 16 + llo;
#pragma unroll
            for (int i = 0; i < 8; ++i) {
                float sA = fmaf(-2.0f, accA[i], ecur);
                if (sA < best[0][i]) { best[0][i] = sA; bidx[0][i] = cand; }
                float sB = fmaf(-2.0f, accB[i], ecur);
                if (sB < best[1][i]) { best[1][i] = sB; bidx[1][i] = cand; }
            }
            b0 = nb0; b1 = nb1; ecur = enxt;
        }

        // ---- branchless argmin across the 16 columns ----
#pragma unroll
        for (int s = 0; s < 2; ++s) {
#pragma unroll
            for (int m = 1; m < 16; m <<= 1) {
#pragma unroll
                for (int i = 0; i < 8; ++i) {
                    float os = __shfl_xor(best[s][i], m, 32);
                    int   oi = __shfl_xor(bidx[s][i], m, 32);
                    int lt = (int)(os < best[s][i]);
                    int eq = (int)(os == best[s][i]);
                    int lo = (int)(oi < bidx[s][i]);
                    int take = lt | (eq & lo);
                    best[s][i] = take ? os : best[s][i];
                    bidx[s][i] = take ? oi : bidx[s][i];
                }
            }
        }

        // ---- gather winning f32 code rows to output + loss ----
        float lsum = 0.f;
#pragma unroll
        for (int s = 0; s < 2; ++s) {
#pragma unroll
            for (int i = 0; i < 8; ++i) {
                const int row = row_base + s * 16 + lhi * 8 + i;
                const int idx = bidx[s][i];
                float zs = __shfl(zsq[s], lhi * 8 + i, 32);
                lsum += best[s][i] + zs;                 // = ||zq - zg||^2
                const float* crow = cb + ((size_t)(g * K_ + idx)) * C_;
                float* orow = out + (size_t)row * D_ + g * C_;
#pragma unroll
                for (int j = 0; j < 4; ++j)
                    orow[j * 16 + llo] = crow[j * 16 + llo];
            }
        }
        if (llo == 0)
            atomicAdd(loss, lsum * (1.5f / ((float)N_ * (float)D_)));
    }
}

// ---------------------------------------------------------------------------
extern "C" void kernel_launch(void* const* d_in, const int* in_sizes, int n_in,
                              void* d_out, int out_size, void* d_ws, size_t ws_size,
                              hipStream_t stream) {
    const float* z  = (const float*)d_in[0];
    const float* cb = (const float*)d_in[1];
    float* out  = (float*)d_out;
    float* loss = out + (size_t)N_ * D_;

    _Float16* cbws = (_Float16*)d_ws;                                   // 512 KB
    float* esqws   = (float*)((char*)d_ws + (size_t)G_ * K_ * C_ * 2);  //  16 KB

    vq_prep<<<dim3(16), dim3(256), 0, stream>>>(cb, cbws, esqws, loss);

    dim3 grid(N_ / ROWS_PER_WG, G_);         // 512 x 4 workgroups
    vq_main<<<grid, dim3(256), SMEM_BYTES, stream>>>(z, cb, cbws, esqws, out, loss);
}
